// IndigoModel_49787260895621
// MI455X (gfx1250) — matrix operand order repaired and checked by reading, not compiled, not run on hardware
//
#include <hip/hip_runtime.h>

// ---------------------------------------------------------------------------
// CDNA5 / gfx1250 transformer forward.
// B=1, S=512, D=1024, H=16, HD=64, DFF=4096, V=32000, L=2.
// All GEMMs run through v_wmma_f32_16x16x32_bf16 (wave32); weights converted
// once to bf16 in workspace (~91MB of operands -> lives in 192MB L2).
// rel-position term reduced to 3 precomputed dots per (h,q) + table lookup.
// GEMM: block 128 thr = 4 waves in 2x2 grid, 32x32 per wave -> per K-step a
// single load clause (2 A + 2 B frags) feeds 4 independent WMMAs.
// ---------------------------------------------------------------------------

#define S_ 512
#define D_ 1024
#define H_ 16
#define HD_ 64
#define DFF_ 4096
#define V_ 32000
#define TD_ 3072   // 3*D
#define EPS_ 1e-5f

typedef __bf16 bf16_t;
typedef bf16_t v16bf __attribute__((ext_vector_type(16)));
typedef float  v8f   __attribute__((ext_vector_type(8)));

union FragAB { v16bf v; uint4 q[2]; };

__device__ inline unsigned short f2bf(float f) {
    unsigned int u = __float_as_uint(f);
    unsigned int r = u + 0x7FFFu + ((u >> 16) & 1u);   // round-to-nearest-even
    return (unsigned short)(r >> 16);
}

// ------------------------------ epilogue ids -------------------------------
enum { EPI_F32 = 0, EPI_BF16 = 1, EPI_BIAS_RELU_BF16 = 2,
       EPI_BIAS_RESID_F32 = 3, EPI_RESID_F32 = 4 };

// ---------------------------------------------------------------------------
// Batched GEMM: C[z][M,N] = A[z][M,K] (row major, bf16) * Bw[z][N,K]^T (bf16)
// Block = 128 threads = 4 waves (2x2); block tile 64x64; wave tile 32x32.
// ---------------------------------------------------------------------------
template <int EPI>
__global__ __launch_bounds__(128)
void gemm_wmma_bf16(const bf16_t* __restrict__ A,
                    const bf16_t* __restrict__ Bw,
                    void* __restrict__ Cout,
                    const float* __restrict__ bias,
                    const float* __restrict__ resid,
                    int M, int N, int K, int ldc,
                    long strideA, long strideB, long strideC)
{
    const int z = blockIdx.z;
    A  += (long)z * strideA;
    Bw += (long)z * strideB;

    const int lane = threadIdx.x & 31;
    const int wave = threadIdx.x >> 5;       // 0..3
    const int wm   = wave >> 1;              // M half of block tile
    const int wn   = wave & 1;               // N half of block tile
    const int l16  = lane & 15;
    const int half = lane >> 4;

    const int m0 = blockIdx.y * 64 + wm * 32;
    const int n0 = blockIdx.x * 64 + wn * 32;

    v8f acc00 = {}, acc01 = {}, acc10 = {}, acc11 = {};

    const bf16_t* arow0 = A  + (long)(m0 + l16) * K;
    const bf16_t* arow1 = arow0 + (long)16 * K;
    const bf16_t* brow0 = Bw + (long)(n0 + l16) * K;
    const bf16_t* brow1 = brow0 + (long)16 * K;

    for (int kk = 0; kk < K; kk += 32) {
        // A fragments (16x32 bf16): lane = row; two 16B chunks per ISA layout
        FragAB a0, a1, b0, b1;
        a0.q[0] = *(const uint4*)(arow0 + kk + half * 8);
        a0.q[1] = *(const uint4*)(arow0 + kk + 16 + half * 8);
        a1.q[0] = *(const uint4*)(arow1 + kk + half * 8);
        a1.q[1] = *(const uint4*)(arow1 + kk + 16 + half * 8);
        // B fragments (32x16 bf16): lane = column; same two 16B chunks as A
        b0.q[0] = *(const uint4*)(brow0 + kk + half * 8);
        b0.q[1] = *(const uint4*)(brow0 + kk + 16 + half * 8);
        b1.q[0] = *(const uint4*)(brow1 + kk + half * 8);
        b1.q[1] = *(const uint4*)(brow1 + kk + 16 + half * 8);

        if (kk + 32 < K) {
            __builtin_prefetch((const void*)(arow0 + kk + 32), 0, 1);
            __builtin_prefetch((const void*)(brow0 + kk + 32), 0, 1);
        }

        // 4 independent accumulator chains (outer product of fragments)
        acc00 = __builtin_amdgcn_wmma_f32_16x16x32_bf16(
            false, a0.v, false, b0.v, (short)0, acc00, false, false);
        acc01 = __builtin_amdgcn_wmma_f32_16x16x32_bf16(
            false, a0.v, false, b1.v, (short)0, acc01, false, false);
        acc10 = __builtin_amdgcn_wmma_f32_16x16x32_bf16(
            false, a1.v, false, b0.v, (short)0, acc10, false, false);
        acc11 = __builtin_amdgcn_wmma_f32_16x16x32_bf16(
            false, a1.v, false, b1.v, (short)0, acc11, false, false);
    }

    // Epilogue: C/D layout -> VGPR r holds row mt + r + 8*half, col = l16
    v8f accs[2][2] = { { acc00, acc01 }, { acc10, acc11 } };
#pragma unroll
    for (int mi = 0; mi < 2; ++mi) {
#pragma unroll
        for (int ni = 0; ni < 2; ++ni) {
            const int col = n0 + ni * 16 + l16;
#pragma unroll
            for (int r = 0; r < 8; ++r) {
                const int row = m0 + mi * 16 + r + half * 8;
                float val = accs[mi][ni][r];
                const long idx = (long)z * strideC + (long)row * ldc + col;
                if (EPI == EPI_F32) {
                    ((float*)Cout)[idx] = val;
                } else if (EPI == EPI_BF16) {
                    ((unsigned short*)Cout)[idx] = f2bf(val);
                } else if (EPI == EPI_BIAS_RELU_BF16) {
                    val += bias[col];
                    val = fmaxf(val, 0.0f);
                    ((unsigned short*)Cout)[idx] = f2bf(val);
                } else if (EPI == EPI_BIAS_RESID_F32) {
                    val += bias[col] + resid[(long)row * ldc + col];
                    ((float*)Cout)[idx] = val;
                } else { // EPI_RESID_F32
                    val += resid[(long)row * ldc + col];
                    ((float*)Cout)[idx] = val;
                }
            }
        }
    }
}

// ---------------------------------------------------------------------------
// LayerNorm: one block per row, output bf16.
// ---------------------------------------------------------------------------
__global__ __launch_bounds__(256)
void layernorm_kernel(const float* __restrict__ x,
                      const float* __restrict__ g,
                      const float* __restrict__ b,
                      unsigned short* __restrict__ out, int D)
{
    const int row = blockIdx.x;
    const int tid = threadIdx.x;
    const float* xr = x + (long)row * D;

    float s = 0.f, s2 = 0.f;
    for (int i = tid; i < D; i += 256) {
        const float v = xr[i];
        s += v; s2 += v * v;
    }
    __shared__ float rs[256], rq[256];
    rs[tid] = s; rq[tid] = s2;
    __syncthreads();
    for (int off = 128; off > 0; off >>= 1) {
        if (tid < off) { rs[tid] += rs[tid + off]; rq[tid] += rq[tid + off]; }
        __syncthreads();
    }
    const float mean = rs[0] / (float)D;
    const float var  = rq[0] / (float)D - mean * mean;
    const float rstd = rsqrtf(var + EPS_);
    for (int i = tid; i < D; i += 256) {
        out[(long)row * D + i] = f2bf((xr[i] - mean) * rstd * g[i] + b[i]);
    }
}

// ---------------------------------------------------------------------------
// Split qkv -> head-major bf16 Q/K (scale folded into Q), V transposed,
// and qrel[h][s][r] = dot(q_scaled, rel_tab[r]).
// ---------------------------------------------------------------------------
__global__ __launch_bounds__(64)
void qkv_pack_kernel(const float* __restrict__ qkv,
                     const float* __restrict__ rel_tab,
                     unsigned short* __restrict__ Qb,
                     unsigned short* __restrict__ Kb,
                     unsigned short* __restrict__ Vt,
                     float* __restrict__ qrel)
{
    const int s = blockIdx.x, h = blockIdx.y, d = threadIdx.x;
    const float scale = 0.125f;                        // 1/sqrt(HD)
    const long base = (long)s * TD_ + h * HD_ + d;
    const float q = qkv[base] * scale;
    const float k = qkv[base + D_];
    const float v = qkv[base + 2 * D_];

    Qb[((long)h * S_ + s) * HD_ + d] = f2bf(q);
    Kb[((long)h * S_ + s) * HD_ + d] = f2bf(k);
    Vt[((long)h * HD_ + d) * S_ + s] = f2bf(v);

    __shared__ float red[3][64];
    for (int r = 0; r < 3; ++r) red[r][d] = q * rel_tab[r * HD_ + d];
    __syncthreads();
    for (int off = 32; off > 0; off >>= 1) {
        if (d < off)
            for (int r = 0; r < 3; ++r) red[r][d] += red[r][d + off];
        __syncthreads();
    }
    if (d < 3) qrel[((long)h * S_ + s) * 3 + d] = red[d][0];
}

// ---------------------------------------------------------------------------
// Softmax row (h,q): logits = scores + qrel[rel+1], causal+pad mask, -> bf16 P
// ---------------------------------------------------------------------------
__global__ __launch_bounds__(256)
void softmax_kernel(const float* __restrict__ scores,
                    const float* __restrict__ qrel,
                    const int* __restrict__ relm,
                    const int* __restrict__ amask,
                    unsigned short* __restrict__ P)
{
    const int q = blockIdx.x, h = blockIdx.y, tid = threadIdx.x;
    const float* srow = scores + ((long)h * S_ + q) * S_;
    const float* qr   = qrel   + ((long)h * S_ + q) * 3;

    float lv[2];
#pragma unroll
    for (int j = 0; j < 2; ++j) {
        const int k = tid + j * 256;
        const int r = relm[(long)k * S_ + q];          // rel_matrix[k][q]
        float l = srow[k] + qr[r + 1];
        const bool ok = (k <= q) && (amask[k] != 0);
        lv[j] = ok ? l : -1e9f;
    }

    __shared__ float red[256];
    float m = fmaxf(lv[0], lv[1]);
    red[tid] = m; __syncthreads();
    for (int off = 128; off > 0; off >>= 1) {
        if (tid < off) red[tid] = fmaxf(red[tid], red[tid + off]);
        __syncthreads();
    }
    const float mx = red[0];
    __syncthreads();

    const float e0 = __expf(lv[0] - mx);
    const float e1 = __expf(lv[1] - mx);
    red[tid] = e0 + e1; __syncthreads();
    for (int off = 128; off > 0; off >>= 1) {
        if (tid < off) red[tid] += red[tid + off];
        __syncthreads();
    }
    const float inv = 1.0f / red[0];

    unsigned short* prow = P + ((long)h * S_ + q) * S_;
    prow[tid]       = f2bf(e0 * inv);
    prow[tid + 256] = f2bf(e1 * inv);
}

// ----------------------------- small utilities -----------------------------
__global__ __launch_bounds__(256)
void f32_to_bf16_kernel(const float* __restrict__ src,
                        unsigned short* __restrict__ dst, long n)
{
    long i = (long)blockIdx.x * 256 + threadIdx.x;
    const long stride = (long)gridDim.x * 256;
    for (; i < n; i += stride) dst[i] = f2bf(src[i]);
}

__global__ __launch_bounds__(256)
void embed_gather_kernel(const int* __restrict__ xt,
                         const float* __restrict__ embed,
                         float* __restrict__ x)
{
    const int s = blockIdx.x;
    const long tok = xt[s];
    for (int d = threadIdx.x; d < D_; d += 256)
        x[(long)s * D_ + d] = embed[tok * D_ + d];
}

__global__ __launch_bounds__(256)
void copy_f32_kernel(const float* __restrict__ src, float* __restrict__ dst, long n)
{
    long i = (long)blockIdx.x * 256 + threadIdx.x;
    const long stride = (long)gridDim.x * 256;
    for (; i < n; i += stride) dst[i] = src[i];
}

// ---------------------------------------------------------------------------
extern "C" void kernel_launch(void* const* d_in, const int* in_sizes, int n_in,
                              void* d_out, int out_size, void* d_ws, size_t ws_size,
                              hipStream_t stream)
{
    (void)in_sizes; (void)n_in; (void)out_size; (void)ws_size;

    const int*   x_t     = (const int*)  d_in[0];
    const int*   amask   = (const int*)  d_in[1];
    const int*   relm    = (const int*)  d_in[2];
    const float* embed   = (const float*)d_in[3];
    const float* qkv_w   = (const float*)d_in[4];
    const float* rel_tab = (const float*)d_in[5];
    const float* o_w     = (const float*)d_in[6];
    const float* w1      = (const float*)d_in[7];
    const float* b1      = (const float*)d_in[8];
    const float* w2      = (const float*)d_in[9];
    const float* b2      = (const float*)d_in[10];
    const float* ln1_g   = (const float*)d_in[11];
    const float* ln1_b   = (const float*)d_in[12];
    const float* ln2_g   = (const float*)d_in[13];
    const float* ln2_b   = (const float*)d_in[14];
    const float* lnf_g   = (const float*)d_in[15];
    const float* lnf_b   = (const float*)d_in[16];
    const float* out_w   = (const float*)d_in[17];

    // ----------------- workspace carve-out (256B aligned) ------------------
    char* ws = (char*)d_ws;
    size_t off = 0;
    auto alloc = [&](size_t bytes) -> void* {
        void* p = ws + off;
        off = (off + bytes + 255) & ~(size_t)255;
        return p;
    };
    unsigned short* qkvw_bf = (unsigned short*)alloc((size_t)TD_ * D_ * 2);
    unsigned short* ow_bf   = (unsigned short*)alloc((size_t)D_ * D_ * 2);
    unsigned short* w1_bf   = (unsigned short*)alloc((size_t)DFF_ * D_ * 2);
    unsigned short* w2_bf   = (unsigned short*)alloc((size_t)D_ * DFF_ * 2);
    unsigned short* outw_bf = (unsigned short*)alloc((size_t)V_ * D_ * 2);
    float*          x       = (float*)alloc((size_t)S_ * D_ * 4);
    unsigned short* hbf     = (unsigned short*)alloc((size_t)S_ * D_ * 2);
    float*          qkv     = (float*)alloc((size_t)S_ * TD_ * 4);
    unsigned short* Qb      = (unsigned short*)alloc((size_t)H_ * S_ * HD_ * 2);
    unsigned short* Kb      = (unsigned short*)alloc((size_t)H_ * S_ * HD_ * 2);
    unsigned short* Vt      = (unsigned short*)alloc((size_t)H_ * HD_ * S_ * 2);
    float*          qrel    = (float*)alloc((size_t)H_ * S_ * 3 * 4);
    float*          scores  = (float*)alloc((size_t)H_ * S_ * S_ * 4);
    unsigned short* Pbf     = (unsigned short*)alloc((size_t)H_ * S_ * S_ * 2);
    unsigned short* Obf     = (unsigned short*)alloc((size_t)S_ * D_ * 2);
    unsigned short* midbf   = (unsigned short*)alloc((size_t)S_ * DFF_ * 2);

    // ----------------- one-time weight conversion to bf16 ------------------
    f32_to_bf16_kernel<<<2048, 256, 0, stream>>>(qkv_w, qkvw_bf, (long)TD_ * D_);
    f32_to_bf16_kernel<<<1024, 256, 0, stream>>>(o_w,   ow_bf,   (long)D_ * D_);
    f32_to_bf16_kernel<<<2048, 256, 0, stream>>>(w1,    w1_bf,   (long)DFF_ * D_);
    f32_to_bf16_kernel<<<2048, 256, 0, stream>>>(w2,    w2_bf,   (long)D_ * DFF_);
    f32_to_bf16_kernel<<<4096, 256, 0, stream>>>(out_w, outw_bf, (long)V_ * D_);

    embed_gather_kernel<<<S_, 256, 0, stream>>>(x_t, embed, x);

    for (int layer = 0; layer < 2; ++layer) {
        // h = LN1(x)
        layernorm_kernel<<<S_, 256, 0, stream>>>(x, ln1_g, ln1_b, hbf, D_);
        // qkv = h @ qkv_w^T   [512 x 3072]
        gemm_wmma_bf16<EPI_F32><<<dim3(TD_ / 64, S_ / 64, 1), 128, 0, stream>>>(
            (const bf16_t*)hbf, (const bf16_t*)qkvw_bf, qkv, nullptr, nullptr,
            S_, TD_, D_, TD_, 0, 0, 0);
        // split/scale/transpose + rel dots
        qkv_pack_kernel<<<dim3(S_, H_), 64, 0, stream>>>(qkv, rel_tab, Qb, Kb, Vt, qrel);
        // scores[h] = Q[h] @ K[h]^T   [512 x 512] x16 heads
        gemm_wmma_bf16<EPI_F32><<<dim3(S_ / 64, S_ / 64, H_), 128, 0, stream>>>(
            (const bf16_t*)Qb, (const bf16_t*)Kb, scores, nullptr, nullptr,
            S_, S_, HD_, S_, (long)S_ * HD_, (long)S_ * HD_, (long)S_ * S_);
        // P = softmax(scores + rel_bias, causal)   -> bf16
        softmax_kernel<<<dim3(S_, H_), 256, 0, stream>>>(scores, qrel, relm, amask, Pbf);
        // O[:, h*64:(h+1)*64] = P[h] @ V[h]        -> bf16, packed [S, D]
        gemm_wmma_bf16<EPI_BF16><<<dim3(1, S_ / 64, H_), 128, 0, stream>>>(
            (const bf16_t*)Pbf, (const bf16_t*)Vt, Obf, nullptr, nullptr,
            S_, HD_, S_, D_, (long)S_ * S_, (long)HD_ * S_, (long)HD_);
        // x = x + O @ o_w^T
        gemm_wmma_bf16<EPI_RESID_F32><<<dim3(D_ / 64, S_ / 64, 1), 128, 0, stream>>>(
            (const bf16_t*)Obf, (const bf16_t*)ow_bf, x, nullptr, x,
            S_, D_, D_, D_, 0, 0, 0);
        // h2 = LN2(x)
        layernorm_kernel<<<S_, 256, 0, stream>>>(x, ln2_g, ln2_b, hbf, D_);
        // mid = relu(h2 @ w1^T + b1)               -> bf16
        gemm_wmma_bf16<EPI_BIAS_RELU_BF16><<<dim3(DFF_ / 64, S_ / 64, 1), 128, 0, stream>>>(
            (const bf16_t*)hbf, (const bf16_t*)w1_bf, midbf, b1, nullptr,
            S_, DFF_, D_, DFF_, 0, 0, 0);
        // x = x + mid @ w2^T + b2
        gemm_wmma_bf16<EPI_BIAS_RESID_F32><<<dim3(D_ / 64, S_ / 64, 1), 128, 0, stream>>>(
            (const bf16_t*)midbf, (const bf16_t*)w2_bf, x, b2, x,
            S_, D_, DFF_, D_, 0, 0, 0);
    }

    // output 0: final x
    copy_f32_kernel<<<512, 256, 0, stream>>>(x, (float*)d_out, (long)S_ * D_);
    // xf = LNf(x); logits = xf @ out_w^T  [512 x 32000]
    layernorm_kernel<<<S_, 256, 0, stream>>>(x, lnf_g, lnf_b, hbf, D_);
    gemm_wmma_bf16<EPI_F32><<<dim3(V_ / 64, S_ / 64, 1), 128, 0, stream>>>(
        (const bf16_t*)hbf, (const bf16_t*)outw_bf,
        (float*)d_out + (long)S_ * D_, nullptr, nullptr,
        S_, V_, D_, V_, 0, 0, 0);
}
